// InstantaneousFrequencyHead_77764677861412
// MI455X (gfx1250) — 8-bit matrix operand order repaired and checked by reading, compiled, not run on hardware
//
#include <hip/hip_runtime.h>
#include <math.h>

typedef __attribute__((ext_vector_type(8))) int v8i;

#define PI_F     3.14159265358979323846f
#define TWO_PI_F 6.28318530717958647692f

static constexpr int BB    = 8;
static constexpr int SS    = 4096;
static constexpr int DIN   = 1024;
static constexpr int DOUT  = 513;
static constexpr int MTOT  = BB * SS;
static constexpr int NPAD  = 544;
static constexpr int NPAIRS = NPAD / 32;
static constexpr int CHAINS = BB * DOUT;
static constexpr int CHUNKS = 16;
static constexpr int CHUNK_LEN = SS / CHUNKS;
static constexpr int WN = DOUT * DIN;

static constexpr size_t OFF_XQ    = 0;
static constexpr size_t OFF_ISX   = OFF_XQ   + (size_t)MTOT * DIN;
static constexpr size_t OFF_WQ    = OFF_ISX  + (size_t)MTOT * 4;
static constexpr size_t OFF_WPART = OFF_WQ   + (size_t)NPAD * DIN;
static constexpr size_t OFF_WC    = OFF_WPART + 128 * 4;
static constexpr size_t OFF_PSUM  = ((OFF_WC + 4 + 255) / 256) * 256;

static __device__ __forceinline__ float fast_tanh(float x)
{
#if __has_builtin(__builtin_amdgcn_tanhf)
    return __builtin_amdgcn_tanhf(x);
#else
    return tanhf(x);
#endif
}

__global__ void __launch_bounds__(256)
quant_x_kernel(const float* __restrict__ x, signed char* __restrict__ xq,
               float* __restrict__ isx)
{
    const int wave  = threadIdx.x >> 5;
    const int lane  = threadIdx.x & 31;
    const int token = blockIdx.x * 8 + wave;
    const float* row = x + (size_t)token * DIN;

    float4 v[8];
    float mx = 0.0f;
#pragma unroll
    for (int i = 0; i < 8; ++i) {
        v[i] = *(const float4*)(row + i * 128 + lane * 4);
        mx = fmaxf(mx, fmaxf(fmaxf(fabsf(v[i].x), fabsf(v[i].y)),
                             fmaxf(fabsf(v[i].z), fabsf(v[i].w))));
    }
#pragma unroll
    for (int off = 16; off > 0; off >>= 1)
        mx = fmaxf(mx, __shfl_xor(mx, off));

    mx = fmaxf(mx, 1e-5f);
    const float scale = 127.0f / mx;

    int* qrow = (int*)(xq + (size_t)token * DIN);
#pragma unroll
    for (int i = 0; i < 8; ++i) {
        int q0 = (int)fminf(fmaxf(rintf(v[i].x * scale), -128.0f), 127.0f);
        int q1 = (int)fminf(fmaxf(rintf(v[i].y * scale), -128.0f), 127.0f);
        int q2 = (int)fminf(fmaxf(rintf(v[i].z * scale), -128.0f), 127.0f);
        int q3 = (int)fminf(fmaxf(rintf(v[i].w * scale), -128.0f), 127.0f);
        qrow[i * 32 + lane] = (q0 & 0xff) | ((q1 & 0xff) << 8) |
                              ((q2 & 0xff) << 16) | ((q3 & 0xff) << 24);
    }
    if (lane == 0) isx[token] = mx / 127.0f;
}

__global__ void __launch_bounds__(256)
wabs_reduce1(const float* __restrict__ W, float* __restrict__ part)
{
    __shared__ float red[256];
    float s = 0.0f;
    for (int i = blockIdx.x * 256 + threadIdx.x; i < WN; i += 128 * 256)
        s += fabsf(W[i]);
    red[threadIdx.x] = s;
    __syncthreads();
#pragma unroll
    for (int off = 128; off > 0; off >>= 1) {
        if (threadIdx.x < off) red[threadIdx.x] += red[threadIdx.x + off];
        __syncthreads();
    }
    if (threadIdx.x == 0) part[blockIdx.x] = red[0];
}

__global__ void __launch_bounds__(128)
wabs_reduce2(const float* __restrict__ part, float* __restrict__ wc)
{
    __shared__ float red[128];
    red[threadIdx.x] = part[threadIdx.x];
    __syncthreads();
#pragma unroll
    for (int off = 64; off > 0; off >>= 1) {
        if (threadIdx.x < off) red[threadIdx.x] += red[threadIdx.x + off];
        __syncthreads();
    }
    if (threadIdx.x == 0)
        wc[0] = fmaxf(red[0] / (float)WN, 1e-5f);
}

__global__ void __launch_bounds__(256)
quant_w_kernel(const float* __restrict__ W, const float* __restrict__ wc,
               signed char* __restrict__ wq)
{
    const int tid = blockIdx.x * 256 + threadIdx.x;
    const int row = tid >> 8;
    const int kg  = tid & 255;
    int packed = 0;
    if (row < DOUT) {
        const float sw = 1.0f / wc[0];
        float4 v = *(const float4*)(W + (size_t)row * DIN + kg * 4);
        int q0 = (int)fminf(fmaxf(rintf(v.x * sw), -1.0f), 1.0f);
        int q1 = (int)fminf(fmaxf(rintf(v.y * sw), -1.0f), 1.0f);
        int q2 = (int)fminf(fmaxf(rintf(v.z * sw), -1.0f), 1.0f);
        int q3 = (int)fminf(fmaxf(rintf(v.w * sw), -1.0f), 1.0f);
        packed = (q0 & 0xff) | ((q1 & 0xff) << 8) |
                 ((q2 & 0xff) << 16) | ((q3 & 0xff) << 24);
    }
    ((int*)wq)[tid] = packed;
}

__global__ void __launch_bounds__(256)
gemm_wmma_kernel(const signed char* __restrict__ xq,
                 const signed char* __restrict__ wq,
                 const float* __restrict__ isx,
                 const float* __restrict__ wc,
                 const float* __restrict__ bias,
                 const float* __restrict__ cw,
                 float* __restrict__ out)
{
    const int wave = threadIdx.x >> 5;
    const int lane = threadIdx.x & 31;
    const int pair = blockIdx.y * 8 + wave;
    if (pair >= NPAIRS) return;

    const int n0    = pair * 32;
    const int m0    = blockIdx.x * 64;
    const int nlane = lane & 15;
    const int hi    = lane >> 4;

    v8i acc[4][2];
#pragma unroll
    for (int s = 0; s < 4; ++s)
#pragma unroll
        for (int t = 0; t < 2; ++t) acc[s][t] = (v8i){0,0,0,0,0,0,0,0};

    const signed char* bptr0 = wq + (size_t)(n0 + nlane) * DIN + hi * 16;
    const signed char* bptr1 = bptr0 + (size_t)16 * DIN;

    for (int k0 = 0; k0 < DIN; k0 += 64) {
        int4 bl0 = *(const int4*)(bptr0 + k0);
        int4 bl1 = *(const int4*)(bptr0 + k0 + 32);
        int4 bl2 = *(const int4*)(bptr1 + k0);
        int4 bl3 = *(const int4*)(bptr1 + k0 + 32);
        v8i bmat0, bmat1;
        bmat0[0] = bl0.x; bmat0[1] = bl0.y; bmat0[2] = bl0.z; bmat0[3] = bl0.w;
        bmat0[4] = bl1.x; bmat0[5] = bl1.y; bmat0[6] = bl1.z; bmat0[7] = bl1.w;
        bmat1[0] = bl2.x; bmat1[1] = bl2.y; bmat1[2] = bl2.z; bmat1[3] = bl2.w;
        bmat1[4] = bl3.x; bmat1[5] = bl3.y; bmat1[6] = bl3.z; bmat1[7] = bl3.w;

#pragma unroll
        for (int sub = 0; sub < 4; ++sub) {
            const signed char* aptr =
                xq + (size_t)(m0 + sub * 16 + nlane) * DIN + k0 + hi * 16;
            int2 a0 = *(const int2*)(aptr);
            int2 a1 = *(const int2*)(aptr + 8);
            int2 a2 = *(const int2*)(aptr + 32);
            int2 a3 = *(const int2*)(aptr + 40);
            v8i amat;
            amat[0] = a0.x; amat[1] = a0.y; amat[2] = a1.x; amat[3] = a1.y;
            amat[4] = a2.x; amat[5] = a2.y; amat[6] = a3.x; amat[7] = a3.y;

            acc[sub][0] = __builtin_amdgcn_wmma_i32_16x16x64_iu8(
                true, amat, true, bmat0, acc[sub][0], false, false);
            acc[sub][1] = __builtin_amdgcn_wmma_i32_16x16x64_iu8(
                true, amat, true, bmat1, acc[sub][1], false, false);
        }
    }

    const float invsw = wc[0];
    const int col0 = n0 + nlane;
    const int col1 = col0 + 16;
    const bool v0 = (col0 < DOUT);
    const bool v1 = (col1 < DOUT);
    const float cw0 = v0 ? cw[col0] : 0.0f;
    const float cw1 = v1 ? cw[col1] : 0.0f;
    const float bv0 = v0 ? bias[col0] : 0.0f;
    const float bv1 = v1 ? bias[col1] : 0.0f;

#pragma unroll
    for (int sub = 0; sub < 4; ++sub) {
        const int rbase = m0 + sub * 16 + hi * 8;
#pragma unroll
        for (int r = 0; r < 8; ++r) {
            const int row = rbase + r;
            const float s = isx[row] * invsw;
            float f0 = (float)acc[sub][0][r] * s + bv0;
            float f1 = (float)acc[sub][1][r] * s + bv1;
            if (v0) out[(size_t)row * DOUT + col0] = fast_tanh(f0) * PI_F * cw0;
            if (v1) out[(size_t)row * DOUT + col1] = fast_tanh(f1) * PI_F * cw1;
        }
    }
}

__global__ void __launch_bounds__(256)
scan_pass1(const float* __restrict__ out, float* __restrict__ psum)
{
    const int chain = blockIdx.x * 256 + threadIdx.x;
    if (chain >= CHAINS) return;
    const int b = chain / DOUT, o = chain % DOUT;
    const int s0 = blockIdx.y * CHUNK_LEN;
    float s = 0.0f;
    const float* p = out + ((size_t)(b * SS + s0) * DOUT) + o;
    for (int i = 0; i < CHUNK_LEN; ++i) s += p[(size_t)i * DOUT];
    psum[(size_t)blockIdx.y * CHAINS + chain] = s;
}

__global__ void __launch_bounds__(256)
scan_pass2(float* __restrict__ psum)
{
    const int chain = blockIdx.x * 256 + threadIdx.x;
    if (chain >= CHAINS) return;
    float run = 0.0f;
#pragma unroll
    for (int c = 0; c < CHUNKS; ++c) {
        float v = psum[(size_t)c * CHAINS + chain];
        psum[(size_t)c * CHAINS + chain] = run;
        run += v;
    }
}

__global__ void __launch_bounds__(256)
scan_pass3(float* __restrict__ out, const float* __restrict__ psum)
{
    const int chain = blockIdx.x * 256 + threadIdx.x;
    if (chain >= CHAINS) return;
    const int b = chain / DOUT, o = chain % DOUT;
    const int s0 = blockIdx.y * CHUNK_LEN;
    float run = psum[(size_t)blockIdx.y * CHAINS + chain];
    float* p = out + ((size_t)(b * SS + s0) * DOUT) + o;
    for (int i = 0; i < CHUNK_LEN; ++i) {
        run += p[(size_t)i * DOUT];
        float w = fmodf(run + PI_F, TWO_PI_F);
        if (w < 0.0f) w += TWO_PI_F;
        p[(size_t)i * DOUT] = w - PI_F;
    }
}

extern "C" void kernel_launch(void* const* d_in, const int* in_sizes, int n_in,
                              void* d_out, int out_size, void* d_ws, size_t ws_size,
                              hipStream_t stream)
{
    (void)in_sizes; (void)n_in; (void)out_size; (void)ws_size;
    const float* x    = (const float*)d_in[0];
    const float* W    = (const float*)d_in[1];
    const float* bias = (const float*)d_in[2];
    const float* cw   = (const float*)d_in[3];
    float* out        = (float*)d_out;

    char* ws = (char*)d_ws;
    signed char* xq   = (signed char*)(ws + OFF_XQ);
    float*       isx  = (float*)(ws + OFF_ISX);
    signed char* wq   = (signed char*)(ws + OFF_WQ);
    float*       part = (float*)(ws + OFF_WPART);
    float*       wc   = (float*)(ws + OFF_WC);
    float*       psum = (float*)(ws + OFF_PSUM);

    quant_x_kernel<<<MTOT / 8, 256, 0, stream>>>(x, xq, isx);
    wabs_reduce1<<<128, 256, 0, stream>>>(W, part);
    wabs_reduce2<<<1, 128, 0, stream>>>(part, wc);
    quant_w_kernel<<<NPAD, 256, 0, stream>>>(W, wc, wq);

    dim3 ggrid(MTOT / 64, (NPAIRS + 7) / 8);
    gemm_wmma_kernel<<<ggrid, 256, 0, stream>>>(xq, wq, isx, wc, bias, cw, out);

    dim3 sgrid((CHAINS + 255) / 256, CHUNKS);
    scan_pass1<<<sgrid, 256, 0, stream>>>(out, psum);
    scan_pass2<<<(CHAINS + 255) / 256, 256, 0, stream>>>(psum);
    scan_pass3<<<sgrid, 256, 0, stream>>>(out, psum);
}
